// DSAttention_64484638982462
// MI455X (gfx1250) — matrix operand order repaired and checked by reading, not compiled, not run on hardware
//
#include <hip/hip_runtime.h>

// ---------------------------------------------------------------------------
// DSAttention for MI455X (gfx1250, wave32, WMMA bf16 -> f32)
//   B=2, L=2048, H=1024, NH=16, HD=64
// 3 kernels: qkv projection GEMM, flash attention (async-to-LDS, double
// buffered), output projection GEMM.  Intermediates (Q,K,V,ctx) in d_ws bf16.
// ---------------------------------------------------------------------------

typedef __attribute__((ext_vector_type(16))) __bf16 v16bf;
typedef __attribute__((ext_vector_type(8)))  float  v8f;

static constexpr int B_  = 2;
static constexpr int L_  = 2048;
static constexpr int H_  = 1024;
static constexpr int NH_ = 16;
static constexpr int HD_ = 64;

__device__ __forceinline__ unsigned short f2bf(float f) {
  unsigned int u = __float_as_uint(f);
  u += 0x7FFFu + ((u >> 16) & 1u);   // round-to-nearest-even
  return (unsigned short)(u >> 16);
}

union FragU { unsigned int u[8]; v16bf v; };

// A-matrix fragment (16x32 bf16, row-major tile, stride in ushorts).
__device__ __forceinline__ v16bf load_frag_a(const unsigned short* base, int stride, int lane) {
  const int m = lane & 15;
  const int h = lane >> 4;
  const unsigned short* row = base + m * stride;
  FragU f;
#pragma unroll
  for (int j = 0; j < 4; ++j) {
    f.u[j]     = *(const unsigned int*)(row + h * 16 + 2 * j);
    f.u[j + 4] = *(const unsigned int*)(row + h * 16 + 8 + 2 * j);
  }
  return f.v;
}

// B-matrix fragment (32x16 bf16) from a tile stored transposed as [n][k].
__device__ __forceinline__ v16bf load_frag_b(const unsigned short* base, int stride, int lane) {
  const int n = lane & 15;
  const int h = lane >> 4;
  const unsigned short* row = base + n * stride + h * 16;
  FragU f;
#pragma unroll
  for (int j = 0; j < 8; ++j) f.u[j] = *(const unsigned int*)(row + 2 * j);
  return f.v;
}

__device__ __forceinline__ v8f wmma_bf16(v16bf a, v16bf b, v8f c) {
  return __builtin_amdgcn_wmma_f32_16x16x32_bf16(false, a, false, b, (short)0, c, false, false);
}

// Async copy 32B/thread (global -> LDS), ASYNCcnt-tracked; IOFFSET applies to
// both the LDS and global addresses per the ISA, so offset:16 covers B128 #2.
__device__ __forceinline__ void async_copy_b128x2(unsigned lds_addr, const void* gptr) {
  unsigned long long ga = (unsigned long long)gptr;
  asm volatile("global_load_async_to_lds_b128 %0, %1, off\n\t"
               "global_load_async_to_lds_b128 %0, %1, off offset:16"
               :: "v"(lds_addr), "v"(ga) : "memory");
}
__device__ __forceinline__ void async_copy_b32(unsigned lds_addr, const void* gptr) {
  unsigned long long ga = (unsigned long long)gptr;
  asm volatile("global_load_async_to_lds_b32 %0, %1, off"
               :: "v"(lds_addr), "v"(ga) : "memory");
}
__device__ __forceinline__ void wait_async0() {
  asm volatile("s_wait_asynccnt 0x0" ::: "memory");
}

// ---------------------------------------------------------------------------
// Kernel 1: Q/K/V projection.  out[n, o] = sum_h X[n,h] * W[o,h] + b[o]
// Block tile 128x64; 8 waves each own a 16x64 strip.  z selects Wq/Wk/Wv.
// Output bf16 in (B, NH, L, HD) head-split layout.
// ---------------------------------------------------------------------------
__global__ __launch_bounds__(256)
void qkv_proj_kernel(const float* __restrict__ X,
                     const float* __restrict__ Wq, const float* __restrict__ bq,
                     const float* __restrict__ Wk, const float* __restrict__ bk,
                     const float* __restrict__ Wv, const float* __restrict__ bv,
                     unsigned short* __restrict__ Qo,
                     unsigned short* __restrict__ Ko,
                     unsigned short* __restrict__ Vo) {
  const int z = blockIdx.z;
  const float* W    = (z == 0) ? Wq : (z == 1) ? Wk : Wv;
  const float* bia  = (z == 0) ? bq : (z == 1) ? bk : bv;
  unsigned short* O = (z == 0) ? Qo : (z == 1) ? Ko : Vo;

  const int r0 = blockIdx.y * 128;
  const int c0 = blockIdx.x * 64;
  const int tid = threadIdx.x;
  const int lane = tid & 31, wave = tid >> 5;

  __shared__ alignas(16) unsigned short As[128][40];  // 80B rows = 5*16B
  __shared__ alignas(16) unsigned short Bs[64][40];

  v8f acc[4] = {};

  for (int k0 = 0; k0 < H_; k0 += 32) {
    __syncthreads();
    {  // stage A: each thread converts 16 f32 -> bf16
      const int ar = tid >> 1, ac = (tid & 1) * 16;
      const float4* s4 = (const float4*)(X + (size_t)(r0 + ar) * H_ + k0 + ac);
#pragma unroll
      for (int q = 0; q < 4; ++q) {
        float4 fv = s4[q];
        unsigned short* d = &As[ar][ac + q * 4];
        d[0] = f2bf(fv.x); d[1] = f2bf(fv.y); d[2] = f2bf(fv.z); d[3] = f2bf(fv.w);
      }
    }
    if (tid < 128) {  // stage B: W rows are output cols -> natural [n][k]
      const int bn = tid >> 1, bc = (tid & 1) * 16;
      const float4* s4 = (const float4*)(W + (size_t)(c0 + bn) * H_ + k0 + bc);
#pragma unroll
      for (int q = 0; q < 4; ++q) {
        float4 fv = s4[q];
        unsigned short* d = &Bs[bn][bc + q * 4];
        d[0] = f2bf(fv.x); d[1] = f2bf(fv.y); d[2] = f2bf(fv.z); d[3] = f2bf(fv.w);
      }
    }
    __syncthreads();

    const v16bf af = load_frag_a(&As[wave * 16][0], 40, lane);
    v16bf bf[4];
#pragma unroll
    for (int t = 0; t < 4; ++t) bf[t] = load_frag_b(&Bs[t * 16][0], 40, lane);
#pragma unroll
    for (int t = 0; t < 4; ++t) acc[t] = wmma_bf16(af, bf[t], acc[t]);
  }

  const int nl = lane & 15, hf = lane >> 4;
#pragma unroll
  for (int t = 0; t < 4; ++t) {
    const int col  = c0 + t * 16 + nl;
    const int head = col >> 6;          // / HD
    const int d    = col & (HD_ - 1);
    const float bval = bia[col];
#pragma unroll
    for (int r = 0; r < 8; ++r) {
      const int grow = r0 + wave * 16 + r + hf * 8;   // C layout row mapping
      const int b = grow >> 11;                       // / L
      const int l = grow & (L_ - 1);
      O[(((size_t)b * NH_ + head) * L_ + l) * HD_ + d] = f2bf(acc[t][r] + bval);
    }
  }
}

// ---------------------------------------------------------------------------
// Kernel 2: flash attention.  One block = 128 query rows of one (b, h).
// scores = (q.k * tau[b] + delta[b,k]) / sqrt(HD); softmax over 2048 keys.
// K tiles + delta staged with GLOBAL_LOAD_ASYNC_TO_LDS, double buffered.
// ---------------------------------------------------------------------------
__global__ __launch_bounds__(256)
void attn_kernel(const unsigned short* __restrict__ Q,
                 const unsigned short* __restrict__ K,
                 const unsigned short* __restrict__ V,
                 const float* __restrict__ tau,
                 const float* __restrict__ delta,
                 unsigned short* __restrict__ ctx) {
  const int b = blockIdx.z, h = blockIdx.y;
  const int q0 = blockIdx.x * 128;
  const int tid = threadIdx.x, lane = tid & 31, wave = tid >> 5;
  const int nl = lane & 15, hf = lane >> 4;
  const float tb = tau[b];

  const unsigned short* Qh = Q + ((size_t)b * NH_ + h) * L_ * HD_;
  const unsigned short* Kh = K + ((size_t)b * NH_ + h) * L_ * HD_;
  const unsigned short* Vh = V + ((size_t)b * NH_ + h) * L_ * HD_;
  const float* db = delta + (size_t)b * L_;

  __shared__ alignas(16) unsigned short Ks[2][64][72];   // [key][d], 144B rows
  __shared__ alignas(16) unsigned short Vts[2][64][72];  // [d][key] (transposed)
  __shared__ alignas(16) unsigned short Ps[8][16][72];   // per-wave P staging
  __shared__ alignas(16) float Ds[2][64];

  const int srow = tid >> 2, scol = (tid & 3) * 16;      // staging coords

  auto stage = [&](int kt2, int buf) {
    // K tile: async global -> LDS, no VGPR round trip (ASYNCcnt tracked)
    async_copy_b128x2((unsigned)(size_t)&Ks[buf][srow][scol],
                      Kh + (size_t)(kt2 + srow) * HD_ + scol);
    // delta slice: async b32
    if (tid < 64)
      async_copy_b32((unsigned)(size_t)&Ds[buf][tid], db + kt2 + tid);
    // V tile transposed at stage time -> Vts[d][key] (regular path)
    const unsigned short* s = Vh + (size_t)(kt2 + srow) * HD_ + scol;
#pragma unroll
    for (int i = 0; i < 16; ++i) Vts[buf][scol + i][srow] = s[i];
  };

  // Q fragments (16 rows x 64) held in registers for the whole kernel.
  const unsigned short* qrow = Qh + (size_t)(q0 + wave * 16) * HD_;
  const v16bf qf0 = load_frag_a(qrow, HD_, lane);
  const v16bf qf1 = load_frag_a(qrow + 32, HD_, lane);

  v8f cacc[4] = {};
  float rmax[8], rsum[8];
#pragma unroll
  for (int r = 0; r < 8; ++r) { rmax[r] = -3.0e38f; rsum[r] = 0.f; }

  stage(0, 0);
  wait_async0();
  __syncthreads();

  for (int kt = 0; kt < L_; kt += 64) {
    const int cur = (kt >> 6) & 1;
    if (kt + 64 < L_) stage(kt + 64, cur ^ 1);   // overlap with compute below

    // ---- S = Q K^T : load all B-fragments first, then WMMA chain
    v16bf kb[8];
#pragma unroll
    for (int t = 0; t < 4; ++t) {
      kb[2 * t]     = load_frag_b(&Ks[cur][t * 16][0], 72, lane);
      kb[2 * t + 1] = load_frag_b(&Ks[cur][t * 16][32], 72, lane);
    }
    v8f sc[4];
#pragma unroll
    for (int t = 0; t < 4; ++t) {
      v8f a = {};
      a = wmma_bf16(qf0, kb[2 * t], a);
      a = wmma_bf16(qf1, kb[2 * t + 1], a);
      sc[t] = a;
    }

    // ---- scale + online softmax (rows live in lane-halves of width 16)
    float newmax[8];
#pragma unroll
    for (int r = 0; r < 8; ++r) newmax[r] = rmax[r];
#pragma unroll
    for (int t = 0; t < 4; ++t) {
      const float dl = Ds[cur][t * 16 + nl];
#pragma unroll
      for (int r = 0; r < 8; ++r) {
        float s = (sc[t][r] * tb + dl) * 0.125f;   // 1/sqrt(64)
        sc[t][r] = s;
        newmax[r] = fmaxf(newmax[r], s);
      }
    }
#pragma unroll
    for (int r = 0; r < 8; ++r)
#pragma unroll
      for (int m = 1; m < 16; m <<= 1)
        newmax[r] = fmaxf(newmax[r], __shfl_xor(newmax[r], m, 16));

    float corr[8], psum[8];
#pragma unroll
    for (int r = 0; r < 8; ++r) {
      corr[r] = __expf(rmax[r] - newmax[r]);
      rmax[r] = newmax[r];
      psum[r] = 0.f;
    }
#pragma unroll
    for (int t = 0; t < 4; ++t)
#pragma unroll
      for (int r = 0; r < 8; ++r) {
        float p = __expf(sc[t][r] - rmax[r]);
        sc[t][r] = p;
        psum[r] += p;
      }
#pragma unroll
    for (int r = 0; r < 8; ++r) {
#pragma unroll
      for (int m = 1; m < 16; m <<= 1) psum[r] += __shfl_xor(psum[r], m, 16);
      rsum[r] = rsum[r] * corr[r] + psum[r];
    }
#pragma unroll
    for (int t = 0; t < 4; ++t)
#pragma unroll
      for (int r = 0; r < 8; ++r) cacc[t][r] *= corr[r];

    // ---- C-layout -> A-fragment layout via per-wave LDS round trip
#pragma unroll
    for (int t = 0; t < 4; ++t)
#pragma unroll
      for (int r = 0; r < 8; ++r)
        Ps[wave][r + hf * 8][t * 16 + nl] = f2bf(sc[t][r]);
    __asm__ volatile("s_wait_dscnt 0" ::: "memory");

    const v16bf pf0 = load_frag_a(&Ps[wave][0][0], 72, lane);
    const v16bf pf1 = load_frag_a(&Ps[wave][0][32], 72, lane);
    v16bf vb[8];
#pragma unroll
    for (int dt = 0; dt < 4; ++dt) {
      vb[2 * dt]     = load_frag_b(&Vts[cur][dt * 16][0], 72, lane);
      vb[2 * dt + 1] = load_frag_b(&Vts[cur][dt * 16][32], 72, lane);
    }
#pragma unroll
    for (int dt = 0; dt < 4; ++dt) {
      cacc[dt] = wmma_bf16(pf0, vb[2 * dt], cacc[dt]);
      cacc[dt] = wmma_bf16(pf1, vb[2 * dt + 1], cacc[dt]);
    }

    wait_async0();       // next-tile async K/delta landed (issuing wave)
    __syncthreads();     // all waves done reading cur + staging visible
  }

  // ---- normalize + write ctx (bf16, (B, L, H) layout: col = h*HD + d)
  float inv[8];
#pragma unroll
  for (int r = 0; r < 8; ++r) inv[r] = __builtin_amdgcn_rcpf(rsum[r]);
#pragma unroll
  for (int dt = 0; dt < 4; ++dt)
#pragma unroll
    for (int r = 0; r < 8; ++r) {
      const int grow = q0 + wave * 16 + r + hf * 8;
      const int col  = h * HD_ + dt * 16 + nl;
      ctx[((size_t)b * L_ + grow) * H_ + col] = f2bf(cacc[dt][r] * inv[r]);
    }
}

// ---------------------------------------------------------------------------
// Kernel 3: output projection.  ctx(bf16) @ Wo^T + bo -> f32 out.
// A tile staged via GLOBAL_LOAD_ASYNC_TO_LDS (already bf16 in workspace).
// ---------------------------------------------------------------------------
__global__ __launch_bounds__(256)
void out_proj_kernel(const unsigned short* __restrict__ Xb,
                     const float* __restrict__ W, const float* __restrict__ bia,
                     float* __restrict__ out) {
  const int r0 = blockIdx.y * 128;
  const int c0 = blockIdx.x * 64;
  const int tid = threadIdx.x;
  const int lane = tid & 31, wave = tid >> 5;

  __shared__ alignas(16) unsigned short As[128][40];
  __shared__ alignas(16) unsigned short Bs[64][40];

  v8f acc[4] = {};

  for (int k0 = 0; k0 < H_; k0 += 32) {
    __syncthreads();
    {  // stage A: async global -> LDS (raw bf16 bytes)
      const int ar = tid >> 1, ac = (tid & 1) * 16;
      async_copy_b128x2((unsigned)(size_t)&As[ar][ac],
                        Xb + (size_t)(r0 + ar) * H_ + k0 + ac);
    }
    if (tid < 128) {  // stage B: f32 weights -> bf16
      const int bn = tid >> 1, bc = (tid & 1) * 16;
      const float4* s4 = (const float4*)(W + (size_t)(c0 + bn) * H_ + k0 + bc);
#pragma unroll
      for (int q = 0; q < 4; ++q) {
        float4 fv = s4[q];
        unsigned short* d = &Bs[bn][bc + q * 4];
        d[0] = f2bf(fv.x); d[1] = f2bf(fv.y); d[2] = f2bf(fv.z); d[3] = f2bf(fv.w);
      }
    }
    wait_async0();
    __syncthreads();

    const v16bf af = load_frag_a(&As[wave * 16][0], 40, lane);
    v16bf bf[4];
#pragma unroll
    for (int t = 0; t < 4; ++t) bf[t] = load_frag_b(&Bs[t * 16][0], 40, lane);
#pragma unroll
    for (int t = 0; t < 4; ++t) acc[t] = wmma_bf16(af, bf[t], acc[t]);
  }

  const int nl = lane & 15, hf = lane >> 4;
#pragma unroll
  for (int t = 0; t < 4; ++t) {
    const int col = c0 + t * 16 + nl;
    const float bval = bia[col];
#pragma unroll
    for (int r = 0; r < 8; ++r) {
      const int grow = r0 + wave * 16 + r + hf * 8;
      out[(size_t)grow * H_ + col] = acc[t][r] + bval;
    }
  }
}

// ---------------------------------------------------------------------------
extern "C" void kernel_launch(void* const* d_in, const int* in_sizes, int n_in,
                              void* d_out, int out_size, void* d_ws, size_t ws_size,
                              hipStream_t stream) {
  (void)in_sizes; (void)n_in; (void)out_size; (void)ws_size;

  const float* hs    = (const float*)d_in[0];
  const float* tau   = (const float*)d_in[1];
  const float* delta = (const float*)d_in[2];
  const float* Wq    = (const float*)d_in[3];
  const float* bq    = (const float*)d_in[4];
  const float* Wk    = (const float*)d_in[5];
  const float* bk    = (const float*)d_in[6];
  const float* Wv    = (const float*)d_in[7];
  const float* bv    = (const float*)d_in[8];
  const float* Wo    = (const float*)d_in[9];
  const float* bo    = (const float*)d_in[10];

  // workspace: Q, K, V (B,NH,L,HD) bf16 + ctx (B,L,H) bf16 = 32 MB total
  const size_t per = (size_t)B_ * NH_ * L_ * HD_;
  unsigned short* Qw = (unsigned short*)d_ws;
  unsigned short* Kw = Qw + per;
  unsigned short* Vw = Kw + per;
  unsigned short* Cw = Vw + per;

  dim3 blk(256);

  dim3 g1(H_ / 64, (B_ * L_) / 128, 3);
  qkv_proj_kernel<<<g1, blk, 0, stream>>>(hs, Wq, bq, Wk, bk, Wv, bv, Qw, Kw, Vw);

  dim3 g2(L_ / 128, NH_, B_);
  attn_kernel<<<g2, blk, 0, stream>>>(Qw, Kw, Vw, tau, delta, Cw);

  dim3 g3(H_ / 64, (B_ * L_) / 128, 1);
  out_proj_kernel<<<g3, blk, 0, stream>>>(Cw, Wo, bo, (float*)d_out);
}
